// WordSATDecoder_11776800326162
// MI455X (gfx1250) — matrix operand order repaired and checked by reading, not compiled, not run on hardware
//
#include <hip/hip_runtime.h>
#include <math.h>

// ---------------------------------------------------------------------------
// Show-Attend-Tell decoder for MI455X (gfx1250, wave32).
// GEMMs (y = x @ W^T) use v_wmma_f32_16x16x32_bf16. Each wave owns a 16-row
// strip and NT=4 consecutive 16-col tiles: one A fragment feeds 4 independent
// WMMA accumulator chains. Loads are phase-separated from WMMAs inside each
// k-step so the compiler can clause the b128 loads and use partial
// s_wait_loadcnt values (load/matrix overlap) instead of full drains.
// ---------------------------------------------------------------------------

#define B_   32
#define T_   60
#define P_   196          // 14*14
#define FEAT 512
#define EMB  256
#define HID  512
#define AFF  512
#define VOCAB 10000
#define XK   1280         // EMB + FEAT + HID  (packed activation row)
#define GN   2048         // 4*HID
#define NT   4            // N-tiles per wave

typedef __attribute__((ext_vector_type(16))) __bf16 v16bf;
typedef __attribute__((ext_vector_type(8)))  __bf16 v8bf;
typedef __attribute__((ext_vector_type(8)))  float  v8f;

__device__ __forceinline__ unsigned short f32_to_bf16(float f) {
    unsigned int x = __float_as_uint(f);
    unsigned int r = x + 0x7FFFu + ((x >> 16) & 1u);   // round-to-nearest-even
    return (unsigned short)(r >> 16);
}

__device__ __forceinline__ float sigmoidf_(float x) {
    return 1.0f / (1.0f + __expf(-x));
}

// ---------------------------------------------------------------------------
// bf16 WMMA GEMM:  C[M,N] (f32) = A[M,K](bf16, lda) @ Bw[N,K](bf16)^T (+bias)
// One wave per (16 x 16*NT) output strip. M,N multiples of 16; K mult of 32.
// ---------------------------------------------------------------------------
__global__ void gemm_bf16_kernel(const unsigned short* __restrict__ Aus, int lda,
                                 const unsigned short* __restrict__ Bus,
                                 const float* __restrict__ bias,
                                 float* __restrict__ C, int ldc,
                                 int M, int N, int K)
{
    const int mtiles = M >> 4, ntiles = N >> 4;
    const int ngroups = (ntiles + NT - 1) / NT;
    const int wave = blockIdx.x * (blockDim.x >> 5) + (threadIdx.x >> 5);
    if (wave >= mtiles * ngroups) return;
    const int mt  = wave / ngroups;
    const int nt0 = (wave % ngroups) * NT;

    const int lane = threadIdx.x & 31;
    const int half = lane >> 4;          // 0: lanes 0-15, 1: lanes 16-31
    const int l16  = lane & 15;

    const __bf16* A  = (const __bf16*)Aus;
    const __bf16* Bw = (const __bf16*)Bus;
    const __bf16* arow = A + (size_t)(mt * 16 + l16) * lda;

    const __bf16* brow[NT];
#pragma unroll
    for (int j = 0; j < NT; ++j) {
        int nt = nt0 + j; if (nt >= ntiles) nt = ntiles - 1;   // clamp (in-bounds dummy)
        brow[j] = Bw + (size_t)(nt * 16 + l16) * K;
    }

    v8f acc[NT] = {};
    for (int k0 = 0; k0 < K; k0 += 32) {
        // ---- load phase: A fragment + all NT B fragments (distinct regs) ----
        // A 16x32 bf16 layout: lanes 0-15 hold K=k0+[0..7] & k0+16+[0..7];
        //                      lanes 16-31 hold K=k0+8+[0..7] & k0+24+[0..7].
        v8bf a_lo = *(const v8bf*)(arow + k0 + half * 8);
        v8bf a_hi = *(const v8bf*)(arow + k0 + 16 + half * 8);
        v8bf b_lo[NT], b_hi[NT];
#pragma unroll
        for (int j = 0; j < NT; ++j) {
            // B 32x16 layout: same K order as A (lanes 0-15 K=k0+[0..7] & k0+16+[0..7]; lanes 16-31 K=k0+8+[0..7] & k0+24+[0..7]).
            b_lo[j] = *(const v8bf*)(brow[j] + k0 + half * 8);
            b_hi[j] = *(const v8bf*)(brow[j] + k0 + 16 + half * 8);
        }
        // ---- compute phase: 4 independent WMMA chains off one A fragment ----
        v16bf a = __builtin_shufflevector(a_lo, a_hi, 0,1,2,3,4,5,6,7,8,9,10,11,12,13,14,15);
#pragma unroll
        for (int j = 0; j < NT; ++j) {
            v16bf b = __builtin_shufflevector(b_lo[j], b_hi[j],
                                              0,1,2,3,4,5,6,7,8,9,10,11,12,13,14,15);
            acc[j] = __builtin_amdgcn_wmma_f32_16x16x32_bf16(false, a, false, b,
                                                             (short)0, acc[j], false, false);
        }
    }

    // C/D 16x16 f32 layout: VGPR r -> row m = r + half*8, col n = l16.
#pragma unroll
    for (int j = 0; j < NT; ++j) {
        const int nt = nt0 + j;
        if (nt >= ntiles) break;
        const int n = nt * 16 + l16;
        const float bv = bias ? bias[n] : 0.0f;
#pragma unroll
        for (int r = 0; r < 8; ++r) {
            const int m = mt * 16 + half * 8 + r;
            C[(size_t)m * ldc + n] = acc[j][r] + bv;
        }
    }
}

// ---------------------------------------------------------------------------
// Init kernels
// ---------------------------------------------------------------------------

// g[b][c] = mean_p cnn[b][c][p]  -> bf16 (for h0/c0 GEMMs)
__global__ void gmean_kernel(const float* __restrict__ cnn, unsigned short* __restrict__ gb)
{
    int i = blockIdx.x * 256 + threadIdx.x;          // i = b*512 + c
    if (i >= B_ * FEAT) return;
    const float* p = cnn + (size_t)i * P_;
    float s = 0.0f;
    for (int k = 0; k < P_; ++k) s += p[k];
    gb[i] = f32_to_bf16(s * (1.0f / (float)P_));
}

// feats_bf16[b][p][c] = bf16(cnn[b][c][p])   (rows = pixels, K = channels)
__global__ void featsT_kernel(const float* __restrict__ cnn, unsigned short* __restrict__ fb)
{
    long long i = (long long)blockIdx.x * 256 + threadIdx.x;
    if (i >= (long long)B_ * P_ * FEAT) return;
    int c = (int)(i & 511);
    long long bp = i >> 9;
    int p = (int)(bp % P_);
    int b = (int)(bp / P_);
    fb[i] = f32_to_bf16(cnn[((size_t)b * FEAT + c) * P_ + p]);
}

// generic f32 -> bf16
__global__ void cvt_kernel(const float* __restrict__ in, unsigned short* __restrict__ out, int n)
{
    int i = blockIdx.x * 256 + threadIdx.x;
    if (i < n) out[i] = f32_to_bf16(in[i]);
}

// fused gate weight  Wcomb[n][k] : k<768 -> Wih[n][k], else Whh[n][k-768]
__global__ void wcomb_kernel(const float* __restrict__ Wih, const float* __restrict__ Whh,
                             unsigned short* __restrict__ out)
{
    int i = blockIdx.x * 256 + threadIdx.x;          // over GN*XK
    if (i >= GN * XK) return;
    int n = i / XK, k = i % XK;
    float v = (k < EMB + FEAT) ? Wih[(size_t)n * (EMB + FEAT) + k]
                               : Whh[(size_t)n * HID + (k - (EMB + FEAT))];
    out[i] = f32_to_bf16(v);
}

__global__ void biascomb_kernel(const float* __restrict__ bih, const float* __restrict__ bhh,
                                float* __restrict__ out)
{
    int i = blockIdx.x * 256 + threadIdx.x;
    if (i < GN) out[i] = bih[i] + bhh[i];
}

// pack h (f32) -> hb (bf16) and into xfull[:, 768:1280]
__global__ void hpack_kernel(const float* __restrict__ h, unsigned short* __restrict__ hb,
                             unsigned short* __restrict__ xfull)
{
    int i = blockIdx.x * 256 + threadIdx.x;
    if (i >= B_ * HID) return;
    int b = i >> 9, j = i & 511;
    unsigned short v = f32_to_bf16(h[i]);
    hb[i] = v;
    xfull[(size_t)b * XK + (EMB + FEAT) + j] = v;
}

// xfull[b][0:256] = bf16(embed_table[captions[b][t]])   (step-0 priming)
__global__ void pack_emb_kernel(const int* __restrict__ captions,
                                const float* __restrict__ table,
                                unsigned short* __restrict__ xfull, int t)
{
    int i = blockIdx.x * 256 + threadIdx.x;          // over B_*EMB
    if (i >= B_ * EMB) return;
    int b = i >> 8, j = i & 255;
    int tok = captions[b * T_ + t];
    xfull[(size_t)b * XK + j] = f32_to_bf16(table[(size_t)tok * EMB + j]);
}

// ---------------------------------------------------------------------------
// Per-step kernels
// ---------------------------------------------------------------------------

// Fused attention: z = tanh(q[b] + kv[b,p]) . wa ; softmax over p ;
// ctx[b][c] = sum_p alpha_p * cnn[b][c][p]  -> bf16 into xfull[:, 256:768]
__global__ void attn_kernel(const float* __restrict__ q,
                            const float* __restrict__ kv,
                            const float* __restrict__ wa,
                            const float* __restrict__ cnn,
                            unsigned short* __restrict__ xfull)
{
    __shared__ float sq[AFF];
    __shared__ float swa[AFF];
    __shared__ float z[P_ + 4];
    __shared__ float red[256];
    const int b = blockIdx.x, tid = threadIdx.x;

    for (int c = tid; c < AFF; c += 256) { sq[c] = q[b * AFF + c]; swa[c] = wa[c]; }
    __syncthreads();

    for (int p = tid; p < P_; p += 256) {
        const float* kvp = kv + ((size_t)(b * P_ + p)) * AFF;
        float s = 0.0f;
#pragma unroll 4
        for (int c = 0; c < AFF; ++c) s += swa[c] * tanhf(sq[c] + kvp[c]);
        z[p] = s;
    }
    __syncthreads();

    // softmax max-reduce
    float m = -3.4e38f;
    for (int p = tid; p < P_; p += 256) m = fmaxf(m, z[p]);
    red[tid] = m; __syncthreads();
    for (int s = 128; s > 0; s >>= 1) { if (tid < s) red[tid] = fmaxf(red[tid], red[tid + s]); __syncthreads(); }
    const float zmax = red[0]; __syncthreads();

    float es = 0.0f;
    for (int p = tid; p < P_; p += 256) { float e = __expf(z[p] - zmax); z[p] = e; es += e; }
    red[tid] = es; __syncthreads();
    for (int s = 128; s > 0; s >>= 1) { if (tid < s) red[tid] += red[tid + s]; __syncthreads(); }
    const float inv = 1.0f / red[0]; __syncthreads();

    // ctx: cnn is [B,512,196] so the p-loop is contiguous
    for (int c = tid; c < FEAT; c += 256) {
        const float* fp = cnn + ((size_t)b * FEAT + c) * P_;
        float acc = 0.0f;
#pragma unroll 4
        for (int p = 0; p < P_; ++p) acc += z[p] * fp[p];
        xfull[(size_t)b * XK + EMB + c] = f32_to_bf16(acc * inv);
    }
}

// LSTM pointwise update; packs new h as bf16 (hb + xfull h-slice) and
// pre-packs the embedding for the NEXT step (removes a launch from the chain).
__global__ void lstm_kernel(const float* __restrict__ gates,
                            float* __restrict__ h, float* __restrict__ c,
                            unsigned short* __restrict__ hb,
                            unsigned short* __restrict__ xfull,
                            const int* __restrict__ captions,
                            const float* __restrict__ table, int next_t)
{
    int i = blockIdx.x * 256 + threadIdx.x;          // over B_*HID
    if (i >= B_ * HID) return;
    int b = i >> 9, j = i & 511;
    const float* gr = gates + (size_t)b * GN;
    float ig = sigmoidf_(gr[j]);
    float fg = sigmoidf_(gr[HID + j]);
    float gg = tanhf(gr[2 * HID + j]);
    float og = sigmoidf_(gr[3 * HID + j]);
    float cn = fg * c[i] + ig * gg;
    float hn = og * tanhf(cn);
    c[i] = cn; h[i] = hn;
    unsigned short v = f32_to_bf16(hn);
    hb[i] = v;
    xfull[(size_t)b * XK + (EMB + FEAT) + j] = v;

    // fused embedding pack for step next_t (threads i < 32*256)
    if (i < B_ * EMB) {
        int b2 = i >> 8, j2 = i & 255;
        int tok = captions[b2 * T_ + next_t];
        xfull[(size_t)b2 * XK + j2] = f32_to_bf16(table[(size_t)tok * EMB + j2]);
    }
}

// ---------------------------------------------------------------------------
// Host launcher
// ---------------------------------------------------------------------------
extern "C" void kernel_launch(void* const* d_in, const int* in_sizes, int n_in,
                              void* d_out, int out_size, void* d_ws, size_t ws_size,
                              hipStream_t stream)
{
    const float* cnn   = (const float*)d_in[0];
    const int*   caps  = (const int*)  d_in[1];
    const float* Wk    = (const float*)d_in[2];
    const float* Wv    = (const float*)d_in[3];
    const float* wa    = (const float*)d_in[4];
    const float* table = (const float*)d_in[5];
    const float* Wh    = (const float*)d_in[6];
    const float* bh    = (const float*)d_in[7];
    const float* Wc    = (const float*)d_in[8];
    const float* bc    = (const float*)d_in[9];
    const float* Wih   = (const float*)d_in[10];
    const float* Whh   = (const float*)d_in[11];
    const float* bih   = (const float*)d_in[12];
    const float* bhh   = (const float*)d_in[13];
    const float* Wfc   = (const float*)d_in[14];
    const float* bfc   = (const float*)d_in[15];
    float* out = (float*)d_out;

    // bump allocator over d_ws
    size_t off = 0;
    auto alloc = [&](size_t bytes) -> void* {
        void* p = (char*)d_ws + off;
        off += (bytes + 255) & ~(size_t)255;
        return p;
    };
    unsigned short* featsB = (unsigned short*)alloc((size_t)B_ * P_ * FEAT * 2);
    float*          kv     = (float*)         alloc((size_t)B_ * P_ * AFF * 4);
    unsigned short* WkB    = (unsigned short*)alloc((size_t)AFF * HID * 2);
    unsigned short* WvB    = (unsigned short*)alloc((size_t)AFF * FEAT * 2);
    unsigned short* WhB    = (unsigned short*)alloc((size_t)HID * FEAT * 2);
    unsigned short* WcB    = (unsigned short*)alloc((size_t)HID * FEAT * 2);
    unsigned short* WcombB = (unsigned short*)alloc((size_t)GN * XK * 2);
    unsigned short* WfcB   = (unsigned short*)alloc((size_t)VOCAB * HID * 2);
    float*          biasC  = (float*)         alloc((size_t)GN * 4);
    unsigned short* gB     = (unsigned short*)alloc((size_t)B_ * FEAT * 2);
    float*          h      = (float*)         alloc((size_t)B_ * HID * 4);
    float*          cst    = (float*)         alloc((size_t)B_ * HID * 4);
    unsigned short* hb     = (unsigned short*)alloc((size_t)B_ * HID * 2);
    float*          q      = (float*)         alloc((size_t)B_ * AFF * 4);
    unsigned short* xfull  = (unsigned short*)alloc((size_t)B_ * XK * 2);
    float*          gates  = (float*)         alloc((size_t)B_ * GN * 4);
    (void)ws_size; (void)in_sizes; (void)n_in; (void)out_size;

    auto gemm = [&](const unsigned short* A, int lda, const unsigned short* Bw,
                    const float* bias, float* C, int ldc, int M, int N, int K) {
        int ngroups = ((N >> 4) + NT - 1) / NT;
        int waves   = (M >> 4) * ngroups;
        int blocks  = (waves + 7) / 8;                // 256 thr = 8 waves/block
        gemm_bf16_kernel<<<blocks, 256, 0, stream>>>(A, lda, Bw, bias, C, ldc, M, N, K);
    };

    // ---- init / precompute ----
    gmean_kernel <<<(B_ * FEAT + 255) / 256, 256, 0, stream>>>(cnn, gB);
    featsT_kernel<<<(int)(((long long)B_ * P_ * FEAT + 255) / 256), 256, 0, stream>>>(cnn, featsB);
    cvt_kernel   <<<(AFF * HID   + 255) / 256, 256, 0, stream>>>(Wk,  WkB,  AFF * HID);
    cvt_kernel   <<<(AFF * FEAT  + 255) / 256, 256, 0, stream>>>(Wv,  WvB,  AFF * FEAT);
    cvt_kernel   <<<(HID * FEAT  + 255) / 256, 256, 0, stream>>>(Wh,  WhB,  HID * FEAT);
    cvt_kernel   <<<(HID * FEAT  + 255) / 256, 256, 0, stream>>>(Wc,  WcB,  HID * FEAT);
    cvt_kernel   <<<(VOCAB * HID + 255) / 256, 256, 0, stream>>>(Wfc, WfcB, VOCAB * HID);
    wcomb_kernel <<<(GN * XK     + 255) / 256, 256, 0, stream>>>(Wih, Whh, WcombB);
    biascomb_kernel<<<(GN + 255) / 256, 256, 0, stream>>>(bih, bhh, biasC);

    // h0 = g@Wh^T + bh ; c0 = g@Wc^T + bc
    gemm(gB, FEAT, WhB, bh, h,   HID, B_, HID, FEAT);
    gemm(gB, FEAT, WcB, bc, cst, HID, B_, HID, FEAT);
    hpack_kernel  <<<(B_ * HID + 255) / 256, 256, 0, stream>>>(h, hb, xfull);
    pack_emb_kernel<<<(B_ * EMB + 255) / 256, 256, 0, stream>>>(caps, table, xfull, 0);

    // kv = feats @ Wv^T   [B*P, AFF]
    gemm(featsB, FEAT, WvB, nullptr, kv, AFF, B_ * P_, AFF, FEAT);

    // ---- decode loop (sequential dependence) ----
    for (int t = 0; t < T_; ++t) {
        // q = h @ Wk^T
        gemm(hb, HID, WkB, nullptr, q, AFF, B_, AFF, HID);
        attn_kernel<<<B_, 256, 0, stream>>>(q, kv, wa, cnn, xfull);
        // gates = [emb|ctx|h] @ [Wih|Whh]^T + (bih+bhh)
        gemm(xfull, XK, WcombB, biasC, gates, GN, B_, GN, XK);
        int next_t = (t + 1 < T_) ? (t + 1) : (T_ - 1);
        lstm_kernel<<<(B_ * HID + 255) / 256, 256, 0, stream>>>(gates, h, cst, hb, xfull,
                                                                caps, table, next_t);
        // logits_t = h @ Wfc^T + bfc  ->  out[:, t, :]
        gemm(hb, HID, WfcB, bfc, out + (size_t)t * VOCAB, T_ * VOCAB, B_, VOCAB, HID);
    }
}
